// LSTM_9792525435077
// MI455X (gfx1250) — compile-verified, matrix operand order checked by reading, not run on hardware
//
#include <hip/hip_runtime.h>
#include <hip/hip_bf16.h>
#include <math.h>

// ---------------- types ----------------
typedef __bf16 v16bf __attribute__((ext_vector_type(16)));
typedef float  v8f   __attribute__((ext_vector_type(8)));

#define B_     256
#define T_     512
#define D_     128
#define H_     512
#define C_     10
#define KTOT   640            // D + H (input projection fused into recurrent GEMM)
#define NCOLS  2048           // 4 gates * H
#define NKCH   (KTOT / 32)    // 20 K-chunks of 32

// ---------------- workspace layout (bytes) ----------------
// WbT : [2048][640] bf16  (column-major weights: W[k][n] at WbT[n*640+k])
// h0  : [256][512] bf16
// h1  : [256][512] bf16
// c   : [256][512] f32
#define WS_WBT   0ull
#define WS_H0    (WS_WBT + 2048ull*640*2)
#define WS_H1    (WS_H0  + 256ull*512*2)
#define WS_C     (WS_H1  + 256ull*512*2)

__device__ __forceinline__ unsigned short f2bf_bits(float f) {
  unsigned int u = __float_as_uint(f);
  u = u + 0x7FFFu + ((u >> 16) & 1u);     // round-to-nearest-even
  return (unsigned short)(u >> 16);
}
__device__ __forceinline__ float bf2f_bits(unsigned short s) {
  return __uint_as_float(((unsigned int)s) << 16);
}
// Low 32 bits of a generic pointer to __shared__ == LDS byte offset (aperture model).
__device__ __forceinline__ unsigned lds_off(const void* p) {
  return (unsigned)(size_t)p;
}

// ---------------- one-time weight pack: fp32 -> bf16, column-major ----------------
__global__ void prep_weights(const float* __restrict__ wgx, const float* __restrict__ wgh,
                             const float* __restrict__ wix, const float* __restrict__ wih,
                             const float* __restrict__ wfx, const float* __restrict__ wfh,
                             const float* __restrict__ wox, const float* __restrict__ woh,
                             unsigned short* __restrict__ WbT) {
  unsigned idx = blockIdx.x * blockDim.x + threadIdx.x;   // over 2048*640
  if (idx >= (unsigned)(NCOLS * KTOT)) return;
  unsigned n    = idx / KTOT;        // global column (gate-major)
  unsigned k    = idx % KTOT;        // 0..639
  unsigned gate = n >> 9;            // 0=g 1=i 2=f 3=o
  unsigned col  = n & 511u;
  const float* wx = (gate == 0) ? wgx : (gate == 1) ? wix : (gate == 2) ? wfx : wox;
  const float* wh = (gate == 0) ? wgh : (gate == 1) ? wih : (gate == 2) ? wfh : woh;
  float v = (k < D_) ? wx[k * H_ + col] : wh[(k - D_) * H_ + col];
  WbT[idx] = f2bf_bits(v);
}

__global__ void zero_state(unsigned short* __restrict__ h0,
                           unsigned short* __restrict__ h1,
                           float* __restrict__ c) {
  unsigned i = blockIdx.x * blockDim.x + threadIdx.x;     // 256*512
  if (i < (unsigned)(B_ * H_)) { h0[i] = 0; h1[i] = 0; c[i] = 0.0f; }
}

// ---------------- per-timestep fused GEMM + gate kernel ----------------
// grid (2, 16): blockIdx.x = 128-row tile of batch, blockIdx.y = 32-col tile of H.
// Block computes [128 x 32] outputs for ALL 4 gates; applies LSTM cell update.
// Double-buffered LDS; weight/h tiles staged with global_load_async_to_lds_b128
// (ASYNCcnt) overlapping the WMMA burst of the current chunk.
__launch_bounds__(256)
__global__ void lstm_step(const float* __restrict__ x, int t,
                          const unsigned short* __restrict__ WbT,
                          const float* __restrict__ bg, const float* __restrict__ bi,
                          const float* __restrict__ bf_, const float* __restrict__ bo,
                          const unsigned short* __restrict__ h_prev,
                          unsigned short* __restrict__ h_next,
                          float* __restrict__ c_state) {
  const int APITCH = 40;  // shorts; 80B row pitch keeps b128 alignment, staggers banks
  __shared__ __align__(16) unsigned short As[2][128 * APITCH];  // A: 128 rows x 32 k
  __shared__ __align__(16) unsigned short Bs[2][128 * APITCH];  // B: 128 cols x 32 k

  const int tid   = threadIdx.x;
  const int wave  = tid >> 5;
  const int lane  = tid & 31;
  const int lhalf = lane >> 4;      // 0: lanes 0-15, 1: lanes 16-31
  const int l15   = lane & 15;
  const int mtile = blockIdx.x;     // rows base mtile*128
  const int ntile = blockIdx.y;     // cols base ntile*32 (within each gate)

  v8f acc[2][4];
  #pragma unroll
  for (int s = 0; s < 2; ++s)
    #pragma unroll
    for (int g = 0; g < 4; ++g) {
      v8f z = {0.f,0.f,0.f,0.f,0.f,0.f,0.f,0.f};
      acc[s][g] = z;
    }

  // cooperative-load mapping: 256 threads -> 128 rows/cols x 2 half-chunks of 16
  const int arow  = tid >> 1;
  const int ahalf = tid & 1;
  const int bcol  = tid >> 1;          // 0..127 = gate*32 + col-in-tile
  const int bhalf = tid & 1;
  const int bgate = bcol >> 5;
  const int bncol = bcol & 31;

  // Stage K-chunk kk into LDS buffer `buf`.
  auto stage = [&](int kk, int buf) {
    const int kbase = kk * 32;
    // ---- A tile [128 rows x 32 k] ----
    {
      const int grow = mtile * 128 + arow;       // batch row
      const int k0   = kbase + ahalf * 16;
      if (kbase < D_) {
        // x slice [B, t, D]: fp32 -> bf16 must pass through VALU
        const float* xp = x + ((size_t)grow * T_ + t) * D_ + k0;
        unsigned short tmp[16];
        #pragma unroll
        for (int j = 0; j < 4; ++j) {
          float4 f = ((const float4*)xp)[j];
          tmp[4*j+0] = f2bf_bits(f.x); tmp[4*j+1] = f2bf_bits(f.y);
          tmp[4*j+2] = f2bf_bits(f.z); tmp[4*j+3] = f2bf_bits(f.w);
        }
        *(uint4*)&As[buf][arow * APITCH + ahalf * 16 + 0] = *(uint4*)&tmp[0];
        *(uint4*)&As[buf][arow * APITCH + ahalf * 16 + 8] = *(uint4*)&tmp[8];
      } else {
        // h_prev is already bf16: direct async global -> LDS (no VGPR round-trip).
        // INST_OFFSET is added to BOTH the LDS and global addresses (ISA 10.7).
        const unsigned short* hp = h_prev + (size_t)grow * H_ + (k0 - D_);
        unsigned la = lds_off(&As[buf][arow * APITCH + ahalf * 16]);
        asm volatile(
            "global_load_async_to_lds_b128 %0, %1, off\n\t"
            "global_load_async_to_lds_b128 %0, %1, off offset:16"
            :: "v"(la), "v"(hp) : "memory");
      }
    }
    // ---- B tile [128 cols x 32 k]: bf16 weights, async -> LDS ----
    {
      const int gn = bgate * H_ + ntile * 32 + bncol;            // global column
      const unsigned short* wp = WbT + (size_t)gn * KTOT + kbase + bhalf * 16;
      if (kk + 1 < NKCH) __builtin_prefetch(wp + 32, 0, 1);
      unsigned lb = lds_off(&Bs[buf][bcol * APITCH + bhalf * 16]);
      asm volatile(
          "global_load_async_to_lds_b128 %0, %1, off\n\t"
          "global_load_async_to_lds_b128 %0, %1, off offset:16"
          :: "v"(lb), "v"(wp) : "memory");
    }
  };

  // ---- software pipeline: prologue fills buffer 0 ----
  stage(0, 0);
  asm volatile("s_wait_asynccnt 0x0" ::: "memory");
  __syncthreads();

  for (int kk = 0; kk < NKCH; ++kk) {
    const int cur = kk & 1;
    if (kk + 1 < NKCH) stage(kk + 1, (kk + 1) & 1);   // overlap next fetch w/ WMMAs

    // ---- A fragment (16x32 bf16): lanes 0-15 K{0..7,16..23}, lanes 16-31 K{8..15,24..31}
    union { uint4 q[2]; v16bf v; } afrag;
    {
      const int row = wave * 16 + l15;
      const int c0  = lhalf * 8;
      afrag.q[0] = *(const uint4*)&As[cur][row * APITCH + c0];
      afrag.q[1] = *(const uint4*)&As[cur][row * APITCH + c0 + 16];
    }

    // ---- 8 WMMAs per chunk: 2 N-subtiles x 4 gates ----
    #pragma unroll
    for (int g = 0; g < 4; ++g) {
      #pragma unroll
      for (int s = 0; s < 2; ++s) {
        union { uint4 q[2]; v16bf v; } bfrag;
        const int col = g * 32 + s * 16 + l15;
        const int c0  = lhalf * 8;
        bfrag.q[0] = *(const uint4*)&Bs[cur][col * APITCH + c0];
        bfrag.q[1] = *(const uint4*)&Bs[cur][col * APITCH + c0 + 16];
        acc[s][g] = __builtin_amdgcn_wmma_f32_16x16x32_bf16(
            false, afrag.v, false, bfrag.v, (short)0, acc[s][g], false, false);
      }
    }

    // drain next-buffer async fills, then rendezvous (compiler adds ds waits)
    asm volatile("s_wait_asynccnt 0x0" ::: "memory");
    __syncthreads();
  }

  // ---- epilogue: biases, activations, cell update ----
  // C/D layout: VGPR e -> M = e + 8*(lane>=16), N = lane&15 (within 16x16 tile)
  #pragma unroll
  for (int s = 0; s < 2; ++s) {
    #pragma unroll
    for (int e = 0; e < 8; ++e) {
      const int m = wave * 16 + e + lhalf * 8;
      const int b = mtile * 128 + m;
      const int n = ntile * 32 + s * 16 + l15;
      float gv = tanhf(acc[s][0][e] + bg[n]);
      float iv = 1.0f / (1.0f + __expf(-(acc[s][1][e] + bi[n])));
      float fv = 1.0f / (1.0f + __expf(-(acc[s][2][e] + bf_[n])));
      float ov = 1.0f / (1.0f + __expf(-(acc[s][3][e] + bo[n])));
      const size_t idx = (size_t)b * H_ + n;
      float cv = gv * iv + c_state[idx] * fv;
      c_state[idx] = cv;
      h_next[idx]  = f2bf_bits(tanhf(cv) * ov);
    }
  }
}

// ---------------- final classifier: out = hT @ wph + bp ----------------
__global__ void classify(const unsigned short* __restrict__ h,
                         const float* __restrict__ wph,
                         const float* __restrict__ bp,
                         float* __restrict__ out) {
  const int b   = blockIdx.x;
  const int tid = threadIdx.x;
  if (tid < C_) {
    float s = bp[tid];
    for (int k = 0; k < H_; ++k)
      s += bf2f_bits(h[(size_t)b * H_ + k]) * wph[k * C_ + tid];
    out[b * C_ + tid] = s;
  }
}

// ---------------- host ----------------
extern "C" void kernel_launch(void* const* d_in, const int* in_sizes, int n_in,
                              void* d_out, int out_size, void* d_ws, size_t ws_size,
                              hipStream_t stream) {
  const float* x   = (const float*)d_in[0];
  const float* wgx = (const float*)d_in[1];
  const float* wgh = (const float*)d_in[2];
  const float* bg  = (const float*)d_in[3];
  const float* wix = (const float*)d_in[4];
  const float* wih = (const float*)d_in[5];
  const float* bi  = (const float*)d_in[6];
  const float* wfx = (const float*)d_in[7];
  const float* wfh = (const float*)d_in[8];
  const float* bf_ = (const float*)d_in[9];
  const float* wox = (const float*)d_in[10];
  const float* woh = (const float*)d_in[11];
  const float* bo  = (const float*)d_in[12];
  const float* wph = (const float*)d_in[13];
  const float* bp  = (const float*)d_in[14];

  char* ws = (char*)d_ws;
  unsigned short* WbT = (unsigned short*)(ws + WS_WBT);
  unsigned short* h0  = (unsigned short*)(ws + WS_H0);
  unsigned short* h1  = (unsigned short*)(ws + WS_H1);
  float*          cst = (float*)(ws + WS_C);

  prep_weights<<<(NCOLS * KTOT + 255) / 256, 256, 0, stream>>>(
      wgx, wgh, wix, wih, wfx, wfh, wox, woh, WbT);
  zero_state<<<(B_ * H_ + 255) / 256, 256, 0, stream>>>(h0, h1, cst);

  unsigned short* hb[2] = {h0, h1};
  for (int t = 0; t < T_; ++t) {
    lstm_step<<<dim3(2, 16), 256, 0, stream>>>(
        x, t, WbT, bg, bi, bf_, bo, hb[t & 1], hb[(t + 1) & 1], cst);
  }
  classify<<<B_, 32, 0, stream>>>(hb[T_ & 1], wph, bp, (float*)d_out);
}
